// graph_constructor2_65498251264079
// MI455X (gfx1250) — matrix operand order repaired and checked by reading, not compiled, not run on hardware
//
#include <hip/hip_runtime.h>
#include <hip/hip_bf16.h>
#include <cstdint>
#include <cstddef>

typedef __attribute__((ext_vector_type(16))) __bf16 v16bf;
typedef __attribute__((ext_vector_type(8)))  __bf16 v8bf;
typedef __attribute__((ext_vector_type(8)))  float  v8f;
typedef __attribute__((ext_vector_type(4)))  float  v4f;

#define NN    1024   // nodes
#define BB    8      // batch
#define FF    8      // feature groups
#define DD    64     // inner dim
#define TOPK  20

// ---------------------------------------------------------------------------
// CDNA5 async global->LDS path (ASYNCcnt), with sync fallback.
// Builtin signature (from compiler diagnostic): first arg is a non-const
// pointer to a 16B int vector in AS1 (global); second is AS3 (LDS).
// ---------------------------------------------------------------------------
#if defined(__has_builtin)
#  if __has_builtin(__builtin_amdgcn_global_load_async_to_lds_b128)
#    define HAVE_ASYNC_LDS 1
#  endif
#endif

#if defined(HAVE_ASYNC_LDS)
typedef int gv4i __attribute__((vector_size(16)));
#  define ASYNC_WAIT() asm volatile("s_wait_asynccnt 0" ::: "memory")
__device__ __forceinline__ void async_copy16(const __bf16* g, __bf16* l)
{
    __builtin_amdgcn_global_load_async_to_lds_b128(
        (__attribute__((address_space(1))) gv4i*)g,
        (__attribute__((address_space(3))) gv4i*)l, 0, 0);
}
#else
#  define ASYNC_WAIT()
#endif

// Copy a ROWS x 32 bf16 panel (row stride = rowStride elems) into LDS,
// 16B per chunk, 256 threads cooperating.
template <int ROWS>
__device__ __forceinline__ void stage_bf16_panel(const __bf16* __restrict__ src,
                                                 size_t rowStride,
                                                 __bf16* dst, int tid)
{
    constexpr int CH = ROWS * 4;          // 16B chunks
    #pragma unroll
    for (int i = 0; i < CH / 256; ++i) {
        const int c = tid + i * 256;
        const int r = c >> 2, s = c & 3;
        const __bf16* g = src + (size_t)r * rowStride + s * 8;
        __bf16*       l = dst + r * 32 + s * 8;
#if defined(HAVE_ASYNC_LDS)
        async_copy16(g, l);
#else
        *(v8bf*)l = *(const v8bf*)g;
#endif
    }
}

// Copy a ROWS x 32 f32 panel into LDS as bf16 (converted in flight).
template <int ROWS>
__device__ __forceinline__ void stage_f32_panel(const float* __restrict__ src,
                                                size_t rowStride,
                                                __bf16* dst, int tid)
{
    constexpr int CH = ROWS * 4;          // 8-float chunks
    #pragma unroll
    for (int i = 0; i < CH / 256; ++i) {
        const int c = tid + i * 256;
        const int r = c >> 2, s = c & 3;
        const float* g = src + (size_t)r * rowStride + s * 8;
        v4f f0 = *(const v4f*)g;
        v4f f1 = *(const v4f*)(g + 4);
        v8bf h;
        #pragma unroll
        for (int j = 0; j < 4; ++j) { h[j] = (__bf16)f0[j]; h[4 + j] = (__bf16)f1[j]; }
        *(v8bf*)(dst + r * 32 + s * 8) = h;
    }
}

// A fragment (16-bit 16x32 layout): lanes 0-15 row M=lane, K {0..7,16..23};
// lanes 16-31 row M=lane-16, K {8..15,24..31}.
__device__ __forceinline__ v16bf frag_a(const __bf16* panel, int blk16, int row, int half)
{
    const __bf16* p = panel + (blk16 * 16 + row) * 32 + half * 8;
    v8bf lo = *(const v8bf*)p;
    v8bf hi = *(const v8bf*)(p + 16);
    return __builtin_shufflevector(lo, hi, 0,1,2,3,4,5,6,7,8,9,10,11,12,13,14,15);
}

// B fragment (16-bit 32x16 layout): lane = column; lanes 0-15: K {0..7,16..23},
// lanes 16-31: K {8..15,24..31} (matches A fragment).
__device__ __forceinline__ v16bf frag_b(const __bf16* panel, int blk16, int row, int half)
{
    const __bf16* p = panel + (blk16 * 16 + row) * 32 + half * 8;
    v8bf lo = *(const v8bf*)p;
    v8bf hi = *(const v8bf*)(p + 16);
    return __builtin_shufflevector(lo, hi, 0,1,2,3,4,5,6,7,8,9,10,11,12,13,14,15);
}

#define WMMA_BF16(A, B, C) \
    __builtin_amdgcn_wmma_f32_16x16x32_bf16(false, (A), false, (B), (short)0, (C), false, false)

// ---------------------------------------------------------------------------
// K1: nv = tanh( (1/F) * sum_f X[b,f] @ X[b,f]^T ), stored bf16.
// Block tile 128x64 (8 waves of 32x32, arranged 4x2), K = F*D = 512 in
// 16 stages of 32, double-buffered through LDS with f32->bf16 conversion.
// ---------------------------------------------------------------------------
__global__ __launch_bounds__(256)
void gram_tanh_kernel(const float* __restrict__ x1, const float* __restrict__ x2,
                      __bf16* __restrict__ nv1, __bf16* __restrict__ nv2)
{
    __shared__ __bf16 sA[2][128 * 32];    // 16 KB
    __shared__ __bf16 sB[2][64 * 32];     //  8 KB

    const int lane = threadIdx.x, wave = threadIdx.y;
    const int tid  = wave * 32 + lane;
    const int wr = wave & 3, wc = wave >> 2;
    const int row = lane & 15, half = lane >> 4;

    const int I  = blockIdx.x;            // 0..7   rows base I*128
    const int J  = blockIdx.y;            // 0..15  cols base J*64
    const int bz = blockIdx.z;            // 0..15
    const int b  = bz & 7;
    const float* __restrict__ x  = (bz < 8) ? x1  : x2;
    __bf16*      __restrict__ nv = (bz < 8) ? nv1 : nv2;
    const size_t bOff = (size_t)b * FF * NN * DD;

    v8f acc[2][2] = {};

    auto stage = [&](int s, int buf) {
        const int kk = s * 32;
        const int f  = kk >> 6;           // which feature slice
        const int c  = (kk >> 5) & 1;     // which 32-chunk of D
        const float* base = x + bOff + (size_t)f * NN * DD + c * 32;
        stage_f32_panel<128>(base + (size_t)(I * 128) * DD, DD, sA[buf], tid);
        stage_f32_panel<64> (base + (size_t)(J * 64)  * DD, DD, sB[buf], tid);
    };

    stage(0, 0);
    __syncthreads();

    const int NST = (FF * DD) / 32;       // 16
    for (int s = 0; s < NST; ++s) {
        const int cur = s & 1;
        if (s + 1 < NST) stage(s + 1, cur ^ 1);

        v16bf af[2], bf[2];
        #pragma unroll
        for (int rt = 0; rt < 2; ++rt) af[rt] = frag_a(sA[cur], wr * 2 + rt, row, half);
        #pragma unroll
        for (int ct = 0; ct < 2; ++ct) bf[ct] = frag_b(sB[cur], wc * 2 + ct, row, half);
        #pragma unroll
        for (int rt = 0; rt < 2; ++rt)
            #pragma unroll
            for (int ct = 0; ct < 2; ++ct)
                acc[rt][ct] = WMMA_BF16(af[rt], bf[ct], acc[rt][ct]);

        __syncthreads();
    }

    // C layout: VGPR r -> M = r + 8*half, N = lane&15
    const float invF = 1.0f / (float)FF;
    #pragma unroll
    for (int rt = 0; rt < 2; ++rt)
        #pragma unroll
        for (int ct = 0; ct < 2; ++ct) {
            const int n0 = I * 128 + wr * 32 + rt * 16 + 8 * half;
            const int m  = J * 64  + wc * 32 + ct * 16 + row;
            #pragma unroll
            for (int r = 0; r < 8; ++r)
                nv[((size_t)b * NN + (n0 + r)) * NN + m] = (__bf16)tanhf(acc[rt][ct][r] * invF);
        }
}

// ---------------------------------------------------------------------------
// K2: adj = nv1 @ nv2^T - nv2 @ nv1^T (f32). Block tile 128x64, 8 waves of
// 32x32 with two accumulator sets. K=1024 in 32 stages, double-buffered
// bf16 panels staged via async global->LDS (ASYNCcnt) when available.
// ---------------------------------------------------------------------------
__global__ __launch_bounds__(256)
void adj_kernel(const __bf16* __restrict__ nv1, const __bf16* __restrict__ nv2,
                float* __restrict__ adj)
{
    __shared__ __bf16 sA1[2][128 * 32], sA2[2][128 * 32];   // 16 KB each
    __shared__ __bf16 sB1[2][64 * 32],  sB2[2][64 * 32];    //  8 KB each

    const int lane = threadIdx.x, wave = threadIdx.y;
    const int tid  = wave * 32 + lane;
    const int wr = wave & 3, wc = wave >> 2;
    const int row = lane & 15, half = lane >> 4;

    const int I = blockIdx.x;             // 0..7   rows base I*128
    const int J = blockIdx.y;             // 0..15  cols base J*64
    const int b = blockIdx.z;
    const size_t bOff = (size_t)b * NN * NN;

    const __bf16* __restrict__ A1 = nv1 + bOff + (size_t)(I * 128) * NN;
    const __bf16* __restrict__ A2 = nv2 + bOff + (size_t)(I * 128) * NN;
    const __bf16* __restrict__ B1 = nv1 + bOff + (size_t)(J * 64)  * NN;
    const __bf16* __restrict__ B2 = nv2 + bOff + (size_t)(J * 64)  * NN;

    v8f acc1[2][2] = {};
    v8f acc2[2][2] = {};

    stage_bf16_panel<128>(A1, NN, sA1[0], tid);
    stage_bf16_panel<128>(A2, NN, sA2[0], tid);
    stage_bf16_panel<64> (B1, NN, sB1[0], tid);
    stage_bf16_panel<64> (B2, NN, sB2[0], tid);
    ASYNC_WAIT();
    __syncthreads();

    const int NST = NN / 32;              // 32
    for (int s = 0; s < NST; ++s) {
        const int cur = s & 1;
        if (s + 1 < NST) {
            const int kk = (s + 1) * 32;
            stage_bf16_panel<128>(A1 + kk, NN, sA1[cur ^ 1], tid);
            stage_bf16_panel<128>(A2 + kk, NN, sA2[cur ^ 1], tid);
            stage_bf16_panel<64> (B1 + kk, NN, sB1[cur ^ 1], tid);
            stage_bf16_panel<64> (B2 + kk, NN, sB2[cur ^ 1], tid);
        }

        v16bf a1[2], a2[2], b1f[2], b2f[2];
        #pragma unroll
        for (int rt = 0; rt < 2; ++rt) {
            a1[rt] = frag_a(sA1[cur], wr * 2 + rt, row, half);
            a2[rt] = frag_a(sA2[cur], wr * 2 + rt, row, half);
        }
        #pragma unroll
        for (int ct = 0; ct < 2; ++ct) {
            b1f[ct] = frag_b(sB1[cur], wc * 2 + ct, row, half);
            b2f[ct] = frag_b(sB2[cur], wc * 2 + ct, row, half);
        }
        #pragma unroll
        for (int rt = 0; rt < 2; ++rt)
            #pragma unroll
            for (int ct = 0; ct < 2; ++ct) {
                acc1[rt][ct] = WMMA_BF16(a1[rt], b2f[ct], acc1[rt][ct]);  // nv1 @ nv2^T
                acc2[rt][ct] = WMMA_BF16(a2[rt], b1f[ct], acc2[rt][ct]);  // nv2 @ nv1^T
            }

        ASYNC_WAIT();
        __syncthreads();
    }

    #pragma unroll
    for (int rt = 0; rt < 2; ++rt)
        #pragma unroll
        for (int ct = 0; ct < 2; ++ct) {
            const int n0 = I * 128 + wr * 32 + rt * 16 + 8 * half;
            const int m  = J * 64  + wc * 32 + ct * 16 + row;
            #pragma unroll
            for (int r = 0; r < 8; ++r)
                adj[bOff + (size_t)(n0 + r) * NN + m] = acc1[rt][ct][r] - acc2[rt][ct][r];
        }
}

// ---------------------------------------------------------------------------
// K3: per-column top-20 (over rows). One wave32 per column; each lane owns
// 32 strided rows; 20 rounds of wave argmax via __shfl_xor; tie-break on
// lower index. Saves (index,value) pairs to workspace.
// ---------------------------------------------------------------------------
__global__ __launch_bounds__(256)
void topk_kernel(const float* __restrict__ adj,
                 int* __restrict__ selIdx, float* __restrict__ selVal)
{
    const int lane = threadIdx.x;
    const int col  = blockIdx.x * 8 + threadIdx.y;   // 0..8191
    const int b    = col >> 10;
    const int m    = col & (NN - 1);

    const float* __restrict__ base = adj + ((size_t)b * NN) * NN + m;
    float v[32];
    #pragma unroll
    for (int i = 0; i < 32; ++i)
        v[i] = base[(size_t)(i * 32 + lane) * NN];

    for (int t = 0; t < TOPK; ++t) {
        float bestv = -3.402823466e38f;
        int   besti = 0x7fffffff;
        #pragma unroll
        for (int i = 0; i < 32; ++i) {
            if (v[i] > bestv) { bestv = v[i]; besti = i * 32 + lane; }
        }
        #pragma unroll
        for (int off = 16; off >= 1; off >>= 1) {
            float ov = __shfl_xor(bestv, off, 32);
            int   oi = __shfl_xor(besti, off, 32);
            if (ov > bestv || (ov == bestv && oi < besti)) { bestv = ov; besti = oi; }
        }
        if (lane == (besti & 31)) v[besti >> 5] = -3.402823466e38f;
        if (lane == 0) {
            selIdx[col * TOPK + t] = besti;
            selVal[col * TOPK + t] = bestv;
        }
    }
}

// ---------------------------------------------------------------------------
// K4: zero the output (coalesced 128-bit stores).
// ---------------------------------------------------------------------------
__global__ __launch_bounds__(256)
void zero_kernel(float4* __restrict__ out)
{
    out[(size_t)blockIdx.x * 256 + threadIdx.x] = make_float4(0.f, 0.f, 0.f, 0.f);
}

// ---------------------------------------------------------------------------
// K5: scatter kept entries: out[b, idx, m] = value.
// ---------------------------------------------------------------------------
__global__ __launch_bounds__(256)
void scatter_kernel(const int* __restrict__ selIdx, const float* __restrict__ selVal,
                    float* __restrict__ out)
{
    const int gid = blockIdx.x * 256 + threadIdx.x;   // 0 .. 8*1024*20-1
    const int col = gid / TOPK;
    const int b   = col >> 10;
    const int m   = col & (NN - 1);
    const int n   = selIdx[gid];
    out[((size_t)b * NN + n) * NN + m] = selVal[gid];
}

// ---------------------------------------------------------------------------
extern "C" void kernel_launch(void* const* d_in, const int* in_sizes, int n_in,
                              void* d_out, int out_size, void* d_ws, size_t ws_size,
                              hipStream_t stream)
{
    // d_in[0] = idx (dead code in the reference), d_in[1]/d_in[2] = features
    const float* tfeat = (const float*)d_in[1];
    const float* dfeat = (const float*)d_in[2];
    float* out = (float*)d_out;

    char* ws = (char*)d_ws;
    const size_t nvBytes = (size_t)BB * NN * NN * sizeof(__bf16);  // 16 MiB each
    __bf16* nv1    = (__bf16*)(ws);
    __bf16* nv2    = (__bf16*)(ws + nvBytes);
    int*    selIdx = (int*)   (ws + 2 * nvBytes);
    float*  selVal = (float*) (ws + 2 * nvBytes + (size_t)BB * NN * TOPK * sizeof(int));

    dim3 blk(32, 8);
    // K1: grid (8 I-tiles of 128, 16 J-tiles of 64, 2 tensors x 8 batches)
    gram_tanh_kernel<<<dim3(8, 16, 16), blk, 0, stream>>>(tfeat, dfeat, nv1, nv2);
    // K2: adj into d_out: grid (8, 16, 8 batches)
    adj_kernel<<<dim3(8, 16, BB), blk, 0, stream>>>(nv1, nv2, out);
    // K3: per-column top-20 -> (index,value) pairs
    topk_kernel<<<dim3(1024), blk, 0, stream>>>(out, selIdx, selVal);
    // K4: zero output
    zero_kernel<<<dim3((BB * NN * NN) / (256 * 4)), dim3(256), 0, stream>>>((float4*)out);
    // K5: scatter kept values
    scatter_kernel<<<dim3((BB * NN * TOPK) / 256), dim3(256), 0, stream>>>(selIdx, selVal, out);
}
